// GNNModel_71511205479202
// MI455X (gfx1250) — matrix operand order repaired and checked by reading, not compiled, not run on hardware
//
#include <hip/hip_runtime.h>
#include <hip/hip_bf16.h>

typedef __attribute__((ext_vector_type(16))) _Float16 v16h;
typedef __attribute__((ext_vector_type(8)))  float    v8f;
typedef __attribute__((ext_vector_type(2)))  _Float16 h2;

struct A8 { h2 e[8]; };   // 8 packed half2 == one v16h A operand

#define HID 32
#define NDIM 8
#define EPB 64          // edges per block iteration (4 waves x 16)
#define KT 9            // 8 data K-tiles + 1 bias K-tile
#define MROWS 288       // 256 data rows + 8 bias rows + 24 zero pad

// ---------------------------------------------------------------------------
// Kernel 0: zero the workspace accumulators (agg, cnt, gsum, gcnt)
// ---------------------------------------------------------------------------
__global__ void zero_ws(float* __restrict__ p, long long n) {
    long long i = (long long)blockIdx.x * blockDim.x + threadIdx.x;
    long long stride = (long long)gridDim.x * blockDim.x;
    for (; i < n; i += stride) p[i] = 0.0f;
}

// ---------------------------------------------------------------------------
// Kernel 1: per-edge  msg = outer(h, x[src]) @ M  via WMMA, scatter-add to agg
//   h = relu(edge_attr @ W1 + b1)                     [E,32]   (f16 in LDS)
//   M = reshape(W2) with b2 rows appended             [288,32] (f16 in LDS,
//       register-resident B fragments per wave)
//   A-tile: 16 edges x 32 K of P[e, k*8+i] = h[e,k] * xs[e,i], built with
//   packed v_pk_mul_f16 (no f32->f16 converts in the hot loop).
// ---------------------------------------------------------------------------
__global__ __launch_bounds__(128)
void edge_kernel(const float* __restrict__ x, const float* __restrict__ ea,
                 const int* __restrict__ src, const int* __restrict__ dst,
                 const float* __restrict__ W1, const float* __restrict__ b1,
                 const float* __restrict__ W2, const float* __restrict__ b2,
                 float* __restrict__ agg, float* __restrict__ cnt,
                 int E) {
    __shared__ _Float16 sM[MROWS * HID];   // 18 KB shared B matrix (f16)
    __shared__ _Float16 sH[EPB * HID];     //  4 KB per-edge hidden h (f16)
    __shared__ _Float16 sXS[EPB * NDIM];   //  1 KB gathered src feats (f16)
    __shared__ int      sDst[EPB];

    const int tid  = threadIdx.x;
    const int lane = tid & 31;
    const int wave = tid >> 5;
    const int n    = lane & 15;        // C/D: N index ; A: M (edge) index
    const int hi   = lane >> 4;        // A layout upper/lower K-half group

    // ---- stage M = [reshape(W2); reshape(b2); 0] as f16 -------------------
    for (int i = tid; i < MROWS * HID; i += 128) {
        int row = i >> 5, o = i & 31;
        float v;
        if (row < 256)       v = W2[(row >> 3) * 256 + (row & 7) * 32 + o];
        else if (row < 264)  v = b2[(row - 256) * 32 + o];
        else                 v = 0.0f;
        sM[i] = (_Float16)v;
    }
    __syncthreads();

    // ---- register-resident B fragments (lane = N, halves = K table order) -
    v16h Bf[KT][2];
#pragma unroll
    for (int t = 0; t < KT; ++t) {
#pragma unroll
        for (int nt = 0; nt < 2; ++nt) {
            const _Float16* p = &sM[(t * 32 + 8 * hi) * HID + nt * 16 + n];
            v16h tmp;
#pragma unroll
            for (int j = 0; j < 16; ++j) tmp[j] = p[((j & 7) + (j >> 3) * 16) * HID];
            Bf[t][nt] = tmp;
        }
    }

    for (long long base = (long long)blockIdx.x * EPB; base < E;
         base += (long long)gridDim.x * EPB) {
        __syncthreads();   // previous iteration's LDS reads complete

        // ---- stage dst + edge counts --------------------------------------
        if (tid < EPB) {
            long long e = base + tid;
            int d = -1;
            if (e < E) { d = dst[e]; atomicAdd(&cnt[d], 1.0f); }
            sDst[tid] = d;
        }
        // ---- stage gathered source features x[src] (f16) ------------------
        for (int i = tid; i < EPB * NDIM; i += 128) {
            int el = i >> 3;
            long long e = base + el;
            float v = 0.0f;
            if (e < E) v = x[(long long)src[e] * NDIM + (i & 7)];
            sXS[i] = (_Float16)v;
        }
        // ---- edge MLP: h = relu(ea @ W1 + b1) -> f16 ----------------------
        {
            int el = tid >> 1, k0 = (tid & 1) * 16;
            long long e = base + el;
            float a0 = 0.f, a1 = 0.f, a2 = 0.f;
            if (e < E) { a0 = ea[e * 3]; a1 = ea[e * 3 + 1]; a2 = ea[e * 3 + 2]; }
#pragma unroll
            for (int k = 0; k < 16; ++k) {
                int kk = k0 + k;
                float v = b1[kk] + a0 * W1[kk] + a1 * W1[32 + kk] + a2 * W1[64 + kk];
                sH[el * HID + kk] = (_Float16)fmaxf(v, 0.0f);
            }
        }
        __syncthreads();

        // ---- per-wave 16-edge tile: packed-f16 A build, 18 WMMAs ----------
        const int eb = wave * 16;
        const _Float16* xsp = &sXS[(eb + n) * NDIM];
        const _Float16* hp  = &sH[(eb + n) * HID];
        h2 xq[4];
#pragma unroll
        for (int i = 0; i < 4; ++i) xq[i] = *(const h2*)(xsp + 2 * i);

        v8f c0 = {}; v8f c1 = {};
#pragma unroll
        for (int t = 0; t < 8; ++t) {
            // A layout: halves j<8 -> K = 8*hi + j ; j>=8 -> K = 16+8*hi+(j-8)
            // global k = t*32 + k32 ; h index = k>>3 ; xs index = j&7
            _Float16 h0 = hp[t * 4 + hi];
            _Float16 h1 = hp[t * 4 + 2 + hi];
            h2 H0 = {h0, h0}, H1 = {h1, h1};
            A8 t8;
            t8.e[0] = H0 * xq[0]; t8.e[1] = H0 * xq[1];
            t8.e[2] = H0 * xq[2]; t8.e[3] = H0 * xq[3];
            t8.e[4] = H1 * xq[0]; t8.e[5] = H1 * xq[1];
            t8.e[6] = H1 * xq[2]; t8.e[7] = H1 * xq[3];
            v16h a = __builtin_bit_cast(v16h, t8);
            c0 = __builtin_amdgcn_wmma_f32_16x16x32_f16(false, a, false, Bf[t][0],
                                                        (short)0, c0, false, false);
            c1 = __builtin_amdgcn_wmma_f32_16x16x32_f16(false, a, false, Bf[t][1],
                                                        (short)0, c1, false, false);
        }
        {   // bias K-tile: P[e, 256+i] = xs[i] (i<8), rows of M hold b2
            h2 z = {(_Float16)0, (_Float16)0};
            A8 t8;
#pragma unroll
            for (int i = 0; i < 4; ++i) t8.e[i] = hi ? z : xq[i];
#pragma unroll
            for (int i = 4; i < 8; ++i) t8.e[i] = z;
            v16h a = __builtin_bit_cast(v16h, t8);
            c0 = __builtin_amdgcn_wmma_f32_16x16x32_f16(false, a, false, Bf[8][0],
                                                        (short)0, c0, false, false);
            c1 = __builtin_amdgcn_wmma_f32_16x16x32_f16(false, a, false, Bf[8][1],
                                                        (short)0, c1, false, false);
        }

        // ---- scatter: C/D layout lane=N, vgpr r = M row (+8 for hi lanes) -
#pragma unroll
        for (int r = 0; r < 8; ++r) {
            int mo = r + hi * 8;
            int d  = sDst[eb + mo];
            if (d >= 0) {
                atomicAdd(&agg[d * HID + n],      c0[r]);
                atomicAdd(&agg[d * HID + 16 + n], c1[r]);
            }
        }
    }
}

// ---------------------------------------------------------------------------
// Kernel 2: h_node = relu(x @ root + agg/max(cnt,1) + conv_bias);
//           scatter-add into per-graph sums. One wave per node (lane = hid).
// ---------------------------------------------------------------------------
__global__ void node_kernel(const float* __restrict__ x, const float* __restrict__ root,
                            const float* __restrict__ cbias,
                            const float* __restrict__ agg, const float* __restrict__ cnt,
                            const int* __restrict__ batch,
                            float* __restrict__ gsum, float* __restrict__ gcnt, int N) {
    long long t = (long long)blockIdx.x * blockDim.x + threadIdx.x;
    int nn = (int)(t >> 5), j = (int)(t & 31);
    if (nn >= N) return;
    float acc = cbias[j];
#pragma unroll
    for (int i = 0; i < NDIM; ++i) acc += x[nn * NDIM + i] * root[i * HID + j];
    acc += agg[nn * HID + j] / fmaxf(cnt[nn], 1.0f);
    acc = fmaxf(acc, 0.0f);
    int b = batch[nn];
    atomicAdd(&gsum[b * HID + j], acc);
    if (j == 0) atomicAdd(&gcnt[b], 1.0f);
}

// ---------------------------------------------------------------------------
// Kernel 3: head MLP — one thread per graph (G = 128). Runs once; keep loops
// rolled so the static instruction count stays small.
// ---------------------------------------------------------------------------
__global__ void head_kernel(const float* __restrict__ gsum, const float* __restrict__ gcnt,
                            const float* __restrict__ emb, const float* __restrict__ ratios,
                            const int* __restrict__ ids,
                            const float* __restrict__ fc0w, const float* __restrict__ fc0b,
                            const float* __restrict__ fc1w, const float* __restrict__ fc1b,
                            const float* __restrict__ fc2w, const float* __restrict__ fc2b,
                            float* __restrict__ out, int G) {
    __shared__ float u[64];
    int tid = threadIdx.x;
    if (tid < 64) {
        float s = 0.0f;
        for (int t = 0; t < 5; ++t) s += emb[ids[t] * 64 + tid] * ratios[t];
        u[tid] = s;
    }
    __syncthreads();
    if (tid >= G) return;
    float c = fmaxf(gcnt[tid], 1.0f);
    float pooled[32];
#pragma clang loop unroll(disable)
    for (int j = 0; j < 32; ++j) pooled[j] = gsum[tid * 32 + j] / c;
    float z0[64];
#pragma clang loop unroll(disable)
    for (int o = 0; o < 64; ++o) {
        float a = fc0b[o];
#pragma clang loop unroll(disable)
        for (int j = 0; j < 32; ++j) a += pooled[j] * fc0w[j * 64 + o];
#pragma clang loop unroll(disable)
        for (int j = 0; j < 64; ++j) a += u[j] * fc0w[(32 + j) * 64 + o];
        z0[o] = fmaxf(a, 0.0f);
    }
    float z1[32];
#pragma clang loop unroll(disable)
    for (int o = 0; o < 32; ++o) {
        float a = fc1b[o];
#pragma clang loop unroll(disable)
        for (int j = 0; j < 64; ++j) a += z0[j] * fc1w[j * 32 + o];
        z1[o] = fmaxf(a, 0.0f);
    }
    float a = fc2b[0];
#pragma clang loop unroll(disable)
    for (int j = 0; j < 32; ++j) a += z1[j] * fc2w[j];
    out[tid] = a;
}

// ---------------------------------------------------------------------------
extern "C" void kernel_launch(void* const* d_in, const int* in_sizes, int n_in,
                              void* d_out, int out_size, void* d_ws, size_t ws_size,
                              hipStream_t stream) {
    const float* x      = (const float*)d_in[0];
    const float* ea     = (const float*)d_in[1];
    const float* ratios = (const float*)d_in[2];
    const int*   eidx   = (const int*)  d_in[3];
    const int*   batch  = (const int*)  d_in[4];
    const int*   ids    = (const int*)  d_in[5];
    const float* W1     = (const float*)d_in[6];
    const float* b1     = (const float*)d_in[7];
    const float* W2     = (const float*)d_in[8];
    const float* b2     = (const float*)d_in[9];
    const float* emb    = (const float*)d_in[10];
    const float* root   = (const float*)d_in[11];
    const float* cbias  = (const float*)d_in[12];
    const float* fc0w   = (const float*)d_in[13];
    const float* fc0b   = (const float*)d_in[14];
    const float* fc1w   = (const float*)d_in[15];
    const float* fc1b   = (const float*)d_in[16];
    const float* fc2w   = (const float*)d_in[17];
    const float* fc2b   = (const float*)d_in[18];

    const int N = in_sizes[0] / NDIM;
    const int E = in_sizes[1] / 3;
    const int G = out_size;            // reference output is [G,1] f32

    float* agg  = (float*)d_ws;                 // [N,32]
    float* cnt  = agg + (size_t)N * HID;        // [N]
    float* gsum = cnt + N;                      // [G,32]
    float* gcnt = gsum + (size_t)G * HID;       // [G]
    long long zn = (long long)N * HID + N + (long long)G * HID + G;

    zero_ws<<<512, 256, 0, stream>>>((float*)d_ws, zn);

    const int* src = eidx;
    const int* dst = eidx + E;
    edge_kernel<<<2048, 128, 0, stream>>>(x, ea, src, dst, W1, b1, W2, b2,
                                          agg, cnt, E);

    long long nthreads = (long long)N * HID;
    node_kernel<<<(int)((nthreads + 255) / 256), 256, 0, stream>>>(
        x, root, cbias, agg, cnt, batch, gsum, gcnt, N);

    head_kernel<<<1, 128, 0, stream>>>(gsum, gcnt, emb, ratios, ids,
                                       fc0w, fc0b, fc1w, fc1b, fc2w, fc2b,
                                       (float*)d_out, G);
}
